// LSTM_LogSoftMax_RNN_15590731285198
// MI455X (gfx1250) — matrix operand order repaired and checked by reading, not compiled, not run on hardware
//
#include <hip/hip_runtime.h>
#include <hip/hip_bf16.h>
#include <math.h>

// ---------------- types for WMMA ----------------
typedef __attribute__((ext_vector_type(16))) __bf16 v16bf;
typedef __attribute__((ext_vector_type(8)))  __bf16 v8bf;
typedef __attribute__((ext_vector_type(8)))  float  v8f;

#define T_SEQ  2048
#define NLAYER 4
#define HID    1024
#define GATES4 4096           // 4*HID
#define VOCAB  50257

__device__ __forceinline__ unsigned short f2bf(float f) {
  unsigned u = __float_as_uint(f);
  u += 0x7fffu + ((u >> 16) & 1u);          // round-to-nearest-even
  return (unsigned short)(u >> 16);
}
__device__ __forceinline__ float bf2f(unsigned short s) {
  return __uint_as_float(((unsigned)s) << 16);
}

// ---------------- device-wide generation barrier ----------------
__device__ __forceinline__ void grid_barrier(unsigned* cnt, unsigned nblocks, unsigned gen) {
  __syncthreads();
  if (threadIdx.x == 0) {
    __threadfence();
    __hip_atomic_fetch_add(cnt, 1u, __ATOMIC_ACQ_REL, __HIP_MEMORY_SCOPE_AGENT);
    const unsigned target = nblocks * gen;
    while (__hip_atomic_load(cnt, __ATOMIC_ACQUIRE, __HIP_MEMORY_SCOPE_AGENT) < target)
      __builtin_amdgcn_s_sleep(1);
  }
  __syncthreads();
}

// ---------------- bf16 WMMA GEMM: C[M,N] = A[M,K] * B[N,K]^T + bias ----------------
// A, B fp32 in memory, converted to bf16 while staging into LDS.
// B is row-major [N][K]  (exactly how Wih / Whh / W_out are stored).
// blockIdx.x = M tile (fast) so all M-blocks of one B-strip run together -> B read
// from HBM once, A stays L2-resident. Double-buffered LDS, NT stores for C.
#define BM 128
#define BN 128
#define BK 32

__global__ __launch_bounds__(256) void gemm_bf16_wmma(
    const float* __restrict__ A, int lda,
    const float* __restrict__ B, int ldb,
    const float* __restrict__ bias1, const float* __restrict__ bias2,
    float* __restrict__ C, int ldc,
    int M, int N, int K)
{
  __shared__ unsigned short As[2][BM * BK];   // [m][k] bf16, ping-pong
  __shared__ unsigned short Bs[2][BN * BK];   // [n][k] bf16 (zero-padded past N)

  const int tid  = threadIdx.x;
  const int lane = tid & 31;
  const int wave = tid >> 5;
  const int wm   = wave & 3;   // 4 waves along M -> 128 rows
  const int wn   = wave >> 2;  // 2 waves along N -> 128 cols
  const int m0   = blockIdx.x * BM;   // M on x (fast) for B-strip reuse
  const int n0   = blockIdx.y * BN;

  v8f acc[2][4] = {};

  const int lrow = tid >> 2;        // 0..63
  const int lcol = (tid & 3) << 3;  // 0,8,16,24

  uint4 ra[2], rb[2];               // staged (converted) tile fragments

  auto load_regs = [&](int k0) {
    #pragma unroll
    for (int g = 0; g < 2; ++g) {
      const int row = lrow + g * 64;
      const float* ap = A + (size_t)(m0 + row) * lda + k0 + lcol;
      float4 a0 = *reinterpret_cast<const float4*>(ap);
      float4 a1 = *reinterpret_cast<const float4*>(ap + 4);
      ra[g].x = (unsigned)f2bf(a0.x) | ((unsigned)f2bf(a0.y) << 16);
      ra[g].y = (unsigned)f2bf(a0.z) | ((unsigned)f2bf(a0.w) << 16);
      ra[g].z = (unsigned)f2bf(a1.x) | ((unsigned)f2bf(a1.y) << 16);
      ra[g].w = (unsigned)f2bf(a1.z) | ((unsigned)f2bf(a1.w) << 16);

      const int n = n0 + row;
      uint4 pb = {0u, 0u, 0u, 0u};
      if (n < N) {
        const float* bp = B + (size_t)n * ldb + k0 + lcol;
        float4 b0 = *reinterpret_cast<const float4*>(bp);
        float4 b1 = *reinterpret_cast<const float4*>(bp + 4);
        pb.x = (unsigned)f2bf(b0.x) | ((unsigned)f2bf(b0.y) << 16);
        pb.y = (unsigned)f2bf(b0.z) | ((unsigned)f2bf(b0.w) << 16);
        pb.z = (unsigned)f2bf(b1.x) | ((unsigned)f2bf(b1.y) << 16);
        pb.w = (unsigned)f2bf(b1.z) | ((unsigned)f2bf(b1.w) << 16);
      }
      rb[g] = pb;
    }
  };
  auto store_lds = [&](int buf) {
    #pragma unroll
    for (int g = 0; g < 2; ++g) {
      const int row = lrow + g * 64;
      *reinterpret_cast<uint4*>(&As[buf][row * BK + lcol]) = ra[g];
      *reinterpret_cast<uint4*>(&Bs[buf][row * BK + lcol]) = rb[g];
    }
  };

  load_regs(0);
  store_lds(0);
  __syncthreads();

  const int lr = lane & 15;
  const int hh = lane >> 4;          // 0 / 1
  const int ka = hh * 16;            // A: K 0-15 vs 16-31
  const int kb = hh * 16;            // B: K 0-15 vs 16-31

  int buf = 0;
  for (int k0 = 0; k0 < K; k0 += BK) {
    const bool has_next = (k0 + BK) < K;
    if (has_next) load_regs(k0 + BK);   // issue global loads early

    // ---- fragments: contiguous k = 16*hh + s for both A and B ----
    v16bf af[2];
    #pragma unroll
    for (int tm = 0; tm < 2; ++tm) {
      const unsigned short* base = &As[buf][(wm * 32 + tm * 16 + lr) * BK];
      v8bf lo  = *reinterpret_cast<const v8bf*>(base + ka);
      v8bf hi8 = *reinterpret_cast<const v8bf*>(base + ka + 8);
      af[tm] = __builtin_shufflevector(lo, hi8, 0,1,2,3,4,5,6,7,8,9,10,11,12,13,14,15);
    }
    v16bf bfr[4];
    #pragma unroll
    for (int tn = 0; tn < 4; ++tn)
      bfr[tn] = *reinterpret_cast<const v16bf*>(&Bs[buf][(wn * 64 + tn * 16 + lr) * BK + kb]);

    #pragma unroll
    for (int tm = 0; tm < 2; ++tm)
      #pragma unroll
      for (int tn = 0; tn < 4; ++tn)
        acc[tm][tn] = __builtin_amdgcn_wmma_f32_16x16x32_bf16(
            false, af[tm], false, bfr[tn], (short)0, acc[tm][tn], false, false);

    if (has_next) {
      store_lds(buf ^ 1);   // convert+store behind the WMMAs
      __syncthreads();
    }
    buf ^= 1;
  }

  // ---- epilogue: C layout (lane<16: M=vgpr, N=lane; lane>=16: M=vgpr+8) ----
  const int rbase = hh ? 8 : 0;
  #pragma unroll
  for (int tn = 0; tn < 4; ++tn) {
    const int col = n0 + wn * 64 + tn * 16 + lr;
    if (col >= N) continue;
    float bsum = 0.f;
    if (bias1) bsum += bias1[col];
    if (bias2) bsum += bias2[col];
    #pragma unroll
    for (int tm = 0; tm < 2; ++tm) {
      const int row0 = m0 + wm * 32 + tm * 16 + rbase;
      #pragma unroll
      for (int r = 0; r < 8; ++r)
        __builtin_nontemporal_store(acc[tm][tn][r] + bsum,
                                    &C[(size_t)(row0 + r) * ldc + col]);
    }
  }
}

// ---------------- fp32 -> bf16 weight conversion ----------------
__global__ void cvt_f32_bf16(const float* __restrict__ s, unsigned short* __restrict__ d, int n) {
  int i = (blockIdx.x * 256 + threadIdx.x) * 4;
  if (i + 3 < n) {
    float4 v = *reinterpret_cast<const float4*>(s + i);
    d[i+0] = f2bf(v.x); d[i+1] = f2bf(v.y); d[i+2] = f2bf(v.z); d[i+3] = f2bf(v.w);
  } else {
    for (; i < n; ++i) d[i] = f2bf(s[i]);
  }
}

// ---------------- per-layer init: h state + sync counter ----------------
__global__ void scan_init(const float* __restrict__ h0, float* __restrict__ h, unsigned* cnt) {
  const int tid = threadIdx.x;
  if (tid == 0) *cnt = 0u;
  for (int i = tid; i < HID; i += 256) h[i] = h0[i];
}

// ---------------- persistent recurrent scan (64 blocks, ONE grid barrier/step) ----
// Block b owns hidden units [b*16, b*16+16). It computes exactly the 64 gate rows
// its slice needs ({n, H+n, 2H+n, 3H+n}) into LDS, keeps c in LDS for the whole
// scan, and only publishes the 16 h values globally -> single barrier per step.
__global__ __launch_bounds__(256) void lstm_scan(
    const float* __restrict__ xw,            // [T][4H], already has x@Wih^T + biases
    const unsigned short* __restrict__ whh,  // bf16 [4H][H]
    float* __restrict__ seq_out,             // [T][H]
    float* __restrict__ h,                   // global h (1024), pre-initialized
    const float* __restrict__ c0,            // initial c for this layer (1024)
    unsigned* __restrict__ cnt)
{
  __shared__ float hs[HID];     // staged h (whole vector)
  __shared__ float gl[64];      // this block's 64 gate pre-activations
  __shared__ float cs[16];      // persistent cell state for the slice
  const int tid   = threadIdx.x;
  const int lane  = tid & 31;
  const int wave  = tid >> 5;
  const int hbase = blockIdx.x * 16;
  const unsigned nb = gridDim.x;
  unsigned gen = 0;

  if (tid < 16) cs[tid] = c0[hbase + tid];

  for (int t = 0; t < T_SEQ; ++t) {
    // stage h into LDS via async DMA (4KB: 256 threads x 16B), ASYNCcnt-tracked
    {
      const int i = tid * 4;
      unsigned ldsoff = (unsigned)(unsigned long long)&hs[i];
      const float* gp = h + i;
      asm volatile("global_load_async_to_lds_b128 %0, %1, off"
                   :: "v"(ldsoff), "v"(gp) : "memory");
      asm volatile("s_wait_asynccnt 0x0" ::: "memory");
    }
    __syncthreads();

    // 64 gate rows for this slice: idx = gate*16 + unit
    const float* xwt = xw + (size_t)t * GATES4;
    for (int rr = 0; rr < 8; ++rr) {
      const int idx = wave * 8 + rr;                       // 0..63
      const int row = (idx >> 4) * HID + hbase + (idx & 15);
      const unsigned short* wp = whh + (size_t)row * HID;
      float acc = 0.f;
      #pragma unroll
      for (int j = 0; j < 16; ++j) {
        const int k = (j * 32 + lane) * 2;                 // coalesced 4B/lane
        const unsigned pk = *reinterpret_cast<const unsigned*>(wp + k);
        acc = fmaf(bf2f((unsigned short)(pk & 0xffffu)), hs[k],
              fmaf(bf2f((unsigned short)(pk >> 16)),     hs[k + 1], acc));
      }
      #pragma unroll
      for (int off = 16; off > 0; off >>= 1) acc += __shfl_xor(acc, off, 32);
      if (lane == 0) gl[idx] = acc + xwt[row];
    }
    __syncthreads();

    // cell update for the 16 owned units (gates never leave LDS)
    if (tid < 16) {
      const float ig = 1.f / (1.f + expf(-gl[tid]));
      const float fg = 1.f / (1.f + expf(-gl[16 + tid]));
      const float gg = tanhf(gl[32 + tid]);
      const float og = 1.f / (1.f + expf(-gl[48 + tid]));
      const float cn = fg * cs[tid] + ig * gg;
      const float hn = og * tanhf(cn);
      cs[tid] = cn;
      h[hbase + tid] = hn;
      seq_out[(size_t)t * HID + hbase + tid] = hn;
    }
    grid_barrier(cnt, nb, ++gen);   // publish h for next step
  }
}

// ---------------- in-place log_softmax, online max+sumexp (2 passes) ----------------
__global__ __launch_bounds__(256) void log_softmax_rows(float* __restrict__ out, int N) {
  __shared__ float rm[256];
  __shared__ float rs[256];
  float* p = out + (size_t)blockIdx.x * N;
  const int tid = threadIdx.x;

  float m = -3.402823466e38f, s = 0.f;
  for (int i = tid; i < N; i += 256) {
    const float x  = p[i];
    const float mn = fmaxf(m, x);
    s = s * expf(m - mn) + expf(x - mn);
    m = mn;
  }
  rm[tid] = m; rs[tid] = s;
  __syncthreads();
  for (int st = 128; st > 0; st >>= 1) {
    if (tid < st) {
      const float m2 = rm[tid + st], s2 = rs[tid + st];
      const float mn = fmaxf(rm[tid], m2);
      rs[tid] = rs[tid] * expf(rm[tid] - mn) + s2 * expf(m2 - mn);
      rm[tid] = mn;
    }
    __syncthreads();
  }
  const float lse = rm[0] + logf(rs[0]);
  for (int i = tid; i < N; i += 256)
    __builtin_nontemporal_store(p[i] - lse, &p[i]);
}

// ---------------- launcher ----------------
extern "C" void kernel_launch(void* const* d_in, const int* in_sizes, int n_in,
                              void* d_out, int out_size, void* d_ws, size_t ws_size,
                              hipStream_t stream) {
  (void)in_sizes; (void)n_in; (void)out_size; (void)ws_size;
  const float* x    = (const float*)d_in[0];   // [2048,1024]
  const float* h0   = (const float*)d_in[1];   // [4,1,1024]
  const float* c0   = (const float*)d_in[2];
  const float* Wih  = (const float*)d_in[3];   // [4,4096,1024]
  const float* Whh  = (const float*)d_in[4];   // [4,4096,1024]
  const float* bih  = (const float*)d_in[5];   // [4,4096]
  const float* bhh  = (const float*)d_in[6];
  const float* Wout = (const float*)d_in[7];   // [50257,1024]
  const float* bout = (const float*)d_in[8];   // [50257]
  float* out = (float*)d_out;                  // [2048,50257]

  // scratch carve (~56 MB)
  char* w = (char*)d_ws;
  float*          xwbuf = (float*)w;          w += (size_t)T_SEQ * GATES4 * 4;
  float*          seqA  = (float*)w;          w += (size_t)T_SEQ * HID * 4;
  float*          seqB  = (float*)w;          w += (size_t)T_SEQ * HID * 4;
  unsigned short* whhb  = (unsigned short*)w; w += (size_t)GATES4 * HID * 2;
  float*          hbuf  = (float*)w;          w += (size_t)HID * 4;
  unsigned*       cnt   = (unsigned*)w;       w += 256;

  const float* seq_in = x;
  float*       seq_out = seqA;
  for (int l = 0; l < NLAYER; ++l) {
    // xw = seq_in @ Wih[l]^T + (bih[l] + bhh[l])    (M tiles on x)
    gemm_bf16_wmma<<<dim3(T_SEQ / BM, GATES4 / BN), 256, 0, stream>>>(
        seq_in, HID, Wih + (size_t)l * GATES4 * HID, HID,
        bih + (size_t)l * GATES4, bhh + (size_t)l * GATES4,
        xwbuf, GATES4, T_SEQ, GATES4, HID);
    // Whh[l] -> bf16 (halves L2 traffic per recurrent step)
    cvt_f32_bf16<<<(GATES4 * HID) / (256 * 4), 256, 0, stream>>>(
        Whh + (size_t)l * GATES4 * HID, whhb, GATES4 * HID);
    scan_init<<<1, 256, 0, stream>>>(h0 + (size_t)l * HID, hbuf, cnt);
    lstm_scan<<<64, 256, 0, stream>>>(xwbuf, whhb, seq_out, hbuf,
                                      c0 + (size_t)l * HID, cnt);
    seq_in  = seq_out;
    seq_out = (seq_out == seqA) ? seqB : seqA;
  }

  // logits = seq @ Wout^T + bout, written straight into d_out (NT stores)
  gemm_bf16_wmma<<<dim3(T_SEQ / BM, (VOCAB + BN - 1) / BN), 256, 0, stream>>>(
      seq_in, HID, Wout, HID, bout, nullptr,
      out, VOCAB, T_SEQ, VOCAB, HID);
  log_softmax_rows<<<T_SEQ, 256, 0, stream>>>(out, VOCAB);
}
